// DSEP_70626442215969
// MI455X (gfx1250) — matrix operand order repaired and checked by reading, not compiled, not run on hardware
//
#include <hip/hip_runtime.h>
#include <cstdint>
#include <cstddef>

typedef __attribute__((ext_vector_type(16))) _Float16 v16h;
typedef __attribute__((ext_vector_type(8)))  float    v8f;
typedef __attribute__((ext_vector_type(4)))  float    f4;

#define WMMA_F16(a, b, c) \
  __builtin_amdgcn_wmma_f32_16x16x32_f16(false, (a), false, (b), (short)0, (c), false, false)

// ---- WMMA fragment loaders (wave32) -------------------------------------
// A fragment: 16x32 f16 tile of row-major A[*, ld], rows [r0, r0+16).
// ISA layout: lanes 0-15 -> M=lane, elems 0..7 = K 0..7, elems 8..15 = K 16..23;
//             lanes 16-31 -> same M, K offset +8.
static __device__ __forceinline__ v16h frag_a(const _Float16* __restrict__ A, int ld,
                                              int r0, int k0, int rmax) {
  const int lane = threadIdx.x & 31;
  int r = r0 + (lane & 15);
  if (r > rmax) r = rmax;                       // clamp; clamped rows never stored
  const _Float16* p = A + (size_t)r * ld + k0 + ((lane >> 4) << 3);
  v16h f;
  ((f4*)&f)[0] = *(const f4*)(p);               // 8 halves: K base..base+7
  ((f4*)&f)[1] = *(const f4*)(p + 16);          // 8 halves: K base+16..base+23
  return f;
}

// B fragment: 32x16 f16 tile; W is row-major [N, ld] (i.e. B^T), lane = N,
// lanes 0-15 hold K k0..k0+7,k0+16..k0+23, lanes 16-31 hold K +8 (ISA table order).
static __device__ __forceinline__ v16h frag_b(const _Float16* __restrict__ W, int ld,
                                              int n0, int k0) {
  const int lane = threadIdx.x & 31;
  const int n = n0 + (lane & 15);
  const _Float16* p = W + (size_t)n * ld + k0 + ((lane >> 4) << 3);
  v16h f;
  ((f4*)&f)[0] = *(const f4*)(p);
  ((f4*)&f)[1] = *(const f4*)(p + 16);
  return f;
}

// ---- GEMM: C[M,N] = A[M,K](f16) * W[N,K]^T(f16) (+bias) (+=C if accum) ---
__global__ __launch_bounds__(128) void gemm_f16(const _Float16* __restrict__ A,
                                                const _Float16* __restrict__ W,
                                                const float* __restrict__ bias,
                                                float* __restrict__ C,
                                                int M, int N, int K, int accum) {
  const int wave = blockIdx.x * (blockDim.x >> 5) + (threadIdx.x >> 5);
  const int row0 = wave << 4;
  if (row0 >= M) return;
  const int lane = threadIdx.x & 31;
  const int rmax = M - 1;
  for (int n0 = 0; n0 < N; n0 += 16) {
    v8f acc = {};
    for (int k0 = 0; k0 < K; k0 += 32) {
      v16h a = frag_a(A, K, row0, k0, rmax);
      v16h b = frag_b(W, K, n0, k0);
      acc = WMMA_F16(a, b, acc);
    }
    const int col   = n0 + (lane & 15);
    const int rbase = row0 + ((lane >> 4) << 3);
    const float bv  = bias ? bias[col] : 0.0f;
    for (int j = 0; j < 8; ++j) {
      const int r = rbase + j;
      if (r < M) {
        const size_t idx = (size_t)r * N + col;
        float v = acc[j] + bv;
        if (accum) v += C[idx];
        C[idx] = v;
      }
    }
  }
}

// ---- Fused GRU: out = (1-z)*n + z*hprev, all 6 gate GEMMs + epilogue -----
__global__ __launch_bounds__(128) void gru_fused(const _Float16* __restrict__ X, int Kx,
                                                 const _Float16* __restrict__ Hf,
                                                 const float* __restrict__ hprev,
                                                 const _Float16* __restrict__ Wx,
                                                 const _Float16* __restrict__ Wh,
                                                 const float* __restrict__ bx,
                                                 const float* __restrict__ bh,
                                                 float* __restrict__ out, int M) {
  const int wave = blockIdx.x * (blockDim.x >> 5) + (threadIdx.x >> 5);
  const int row0 = wave << 4;
  if (row0 >= M) return;
  const int lane = threadIdx.x & 31;
  const int rmax = M - 1;
  for (int n0 = 0; n0 < 256; n0 += 16) {
    v8f axr = {}, axz = {}, axn = {};
    v8f ahr = {}, ahz = {}, ahn = {};
    for (int k0 = 0; k0 < Kx; k0 += 32) {
      v16h a = frag_a(X, Kx, row0, k0, rmax);
      axr = WMMA_F16(a, frag_b(Wx, Kx, n0,        k0), axr);
      axz = WMMA_F16(a, frag_b(Wx, Kx, n0 + 256,  k0), axz);
      axn = WMMA_F16(a, frag_b(Wx, Kx, n0 + 512,  k0), axn);
    }
    for (int k0 = 0; k0 < 256; k0 += 32) {
      v16h a = frag_a(Hf, 256, row0, k0, rmax);
      ahr = WMMA_F16(a, frag_b(Wh, 256, n0,       k0), ahr);
      ahz = WMMA_F16(a, frag_b(Wh, 256, n0 + 256, k0), ahz);
      ahn = WMMA_F16(a, frag_b(Wh, 256, n0 + 512, k0), ahn);
    }
    const int col   = n0 + (lane & 15);
    const int rbase = row0 + ((lane >> 4) << 3);
    const float bxr = bx[col], bxz = bx[col + 256], bxn = bx[col + 512];
    const float bhr = bh[col], bhz = bh[col + 256], bhn = bh[col + 512];
    for (int j = 0; j < 8; ++j) {
      const int r = rbase + j;
      if (r >= M) continue;
      const float rr = 1.0f / (1.0f + __expf(-((axr[j] + bxr) + (ahr[j] + bhr))));
      const float zz = 1.0f / (1.0f + __expf(-((axz[j] + bxz) + (ahz[j] + bhz))));
      const float nn = tanhf((axn[j] + bxn) + rr * (ahn[j] + bhn));
      const float hp = hprev[(size_t)r * 256 + col];
      out[(size_t)r * 256 + col] = (1.0f - zz) * nn + zz * hp;
    }
  }
}

// ---- Row L2-normalize (optional eval-mode rrelu first), f32 + f16 out ----
__global__ __launch_bounds__(256) void rownorm(const float* __restrict__ in,
                                               float* __restrict__ outf,
                                               _Float16* __restrict__ outh,
                                               int doRrelu) {
  __shared__ float red[256];
  const int row = blockIdx.x;
  const int c = threadIdx.x;
  float v = in[(size_t)row * 256 + c];
  if (doRrelu && v < 0.0f) v *= 0.2291666666666667f;   // (1/8 + 1/3)/2
  red[c] = v * v;
  __syncthreads();
  for (int s = 128; s > 0; s >>= 1) {
    if (c < s) red[c] += red[c + s];
    __syncthreads();
  }
  const float d = fmaxf(sqrtf(red[0]), 1e-12f);
  const float o = v / d;
  if (outf) outf[(size_t)row * 256 + c] = o;
  if (outh) outh[(size_t)row * 256 + c] = (_Float16)o;
}

// ---- BatchNorm statistics (train-mode, biased var) -----------------------
__global__ __launch_bounds__(256) void bn_partial(const float* __restrict__ x,
                                                  float* __restrict__ s,
                                                  float* __restrict__ s2, int M) {
  const int c = threadIdx.x;
  const int r0 = blockIdx.x * 256;
  float a = 0.0f, b = 0.0f;
  for (int i = 0; i < 256; ++i) {
    const int r = r0 + i;
    if (r < M) {
      const float v = x[(size_t)r * 256 + c];
      a += v; b += v * v;
    }
  }
  atomicAdd(&s[c], a);
  atomicAdd(&s2[c], b);
}

__global__ __launch_bounds__(256) void bn_finalize(float* __restrict__ s,
                                                   float* __restrict__ s2, float invM) {
  const int c = threadIdx.x;
  const float mu = s[c] * invM;
  const float var = s2[c] * invM - mu * mu;
  s[c] = mu;
  s2[c] = rsqrtf(var + 1e-5f);     // store inv-std
}

__global__ __launch_bounds__(256) void bn_apply(const float* __restrict__ x,
                                                const float* __restrict__ mu,
                                                const float* __restrict__ istd,
                                                const float* __restrict__ g,
                                                const float* __restrict__ be,
                                                _Float16* __restrict__ y, int n) {
  const int i = blockIdx.x * 256 + threadIdx.x;
  if (i >= n) return;
  const int c = i & 255;
  y[i] = (_Float16)(((x[i] - mu[c]) * istd[c]) * g[c] + be[c]);
}

// ---- Small utility kernels ----------------------------------------------
__global__ __launch_bounds__(256) void zero_f32(float* __restrict__ p, int n) {
  const int i = blockIdx.x * 256 + threadIdx.x;
  if (i < n) p[i] = 0.0f;
}

__global__ __launch_bounds__(256) void copy_f32(float* __restrict__ d,
                                                const float* __restrict__ s, int n) {
  const int i = blockIdx.x * 256 + threadIdx.x;
  if (i < n) d[i] = s[i];
}

__global__ __launch_bounds__(256) void conv_f16(const float* __restrict__ x,
                                                _Float16* __restrict__ y, int n) {
  const int i = blockIdx.x * 256 + threadIdx.x;
  if (i < n) y[i] = (_Float16)x[i];
}

// x: [R, C] f32 row-major  ->  y: [C, R] f16 row-major (transpose+convert)
__global__ __launch_bounds__(256) void transpose_f16(const float* __restrict__ x,
                                                     _Float16* __restrict__ y,
                                                     int R, int C) {
  const int i = blockIdx.x * 256 + threadIdx.x;
  if (i >= R * C) return;
  const int r = i / C, c = i % C;
  y[(size_t)c * R + r] = (_Float16)x[i];
}

// relcat = concat([rel_emb(230), r_rel(230)]) rows, to f16 (n = 460*256)
__global__ __launch_bounds__(256) void concat_rel(const float* __restrict__ a,
                                                  const float* __restrict__ b,
                                                  _Float16* __restrict__ y, int n) {
  const int i = blockIdx.x * 256 + threadIdx.x;
  if (i >= n) return;
  const int r = i >> 8;
  const float v = (r < 230) ? a[i] : b[i - 230 * 256];
  y[i] = (_Float16)v;
}

// x_input = concat([rel, segmean], axis=1) -> f16 [R2, 512]
__global__ __launch_bounds__(256) void build_xcat(const float* __restrict__ rel,
                                                  const float* __restrict__ sums,
                                                  const float* __restrict__ cnt,
                                                  _Float16* __restrict__ y, int n) {
  const int i = blockIdx.x * 256 + threadIdx.x;
  if (i >= n) return;
  const int r = i >> 9;
  const int c = i & 511;
  const float v = (c < 256) ? rel[r * 256 + c]
                            : sums[r * 256 + (c - 256)] / fmaxf(cnt[r], 1.0f);
  y[i] = (_Float16)v;
}

// segment-sum of h[r_to_e] by rel_seg; thread = (l, 4-col chunk)
__global__ __launch_bounds__(256) void seg_scatter(const float* __restrict__ h,
                                                   const int* __restrict__ rte,
                                                   const int* __restrict__ seg,
                                                   float* __restrict__ sums, int L) {
  const int i = blockIdx.x * 256 + threadIdx.x;
  if (i >= L * 64) return;
  const int l = i >> 6;
  const int c = (i & 63) << 2;
  const float* hp = h + (size_t)rte[l] * 256 + c;
  float* sp = sums + (size_t)seg[l] * 256 + c;
  atomicAdd(sp + 0, hp[0]);
  atomicAdd(sp + 1, hp[1]);
  atomicAdd(sp + 2, hp[2]);
  atomicAdd(sp + 3, hp[3]);
}

__global__ __launch_bounds__(256) void cnt_kernel(const int* __restrict__ seg,
                                                  float* __restrict__ cnt, int L) {
  const int l = blockIdx.x * 256 + threadIdx.x;
  if (l < L) atomicAdd(&cnt[seg[l]], 1.0f);
}

// msg = h[src] + h0[etype]; scatter-add into agg[dst]
__global__ __launch_bounds__(256) void edge_scatter(const float* __restrict__ h,
                                                    const float* __restrict__ h0,
                                                    const int* __restrict__ src,
                                                    const int* __restrict__ dst,
                                                    const int* __restrict__ et,
                                                    float* __restrict__ agg, int EG) {
  const int i = blockIdx.x * 256 + threadIdx.x;
  if (i >= EG * 64) return;
  const int e = i >> 6;
  const int c = (i & 63) << 2;
  const float* hp = h  + (size_t)src[e] * 256 + c;
  const float* rp = h0 + (size_t)et[e]  * 256 + c;
  float* ap = agg + (size_t)dst[e] * 256 + c;
  atomicAdd(ap + 0, hp[0] + rp[0]);
  atomicAdd(ap + 1, hp[1] + rp[1]);
  atomicAdd(ap + 2, hp[2] + rp[2]);
  atomicAdd(ap + 3, hp[3] + rp[3]);
}

__global__ __launch_bounds__(256) void deg_kernel(const int* __restrict__ dst,
                                                  float* __restrict__ deg, int EG) {
  const int e = blockIdx.x * 256 + threadIdx.x;
  if (e < EG) atomicAdd(&deg[dst[e]], 1.0f);
}

// agg / max(deg, 1) -> f16 for WMMA
__global__ __launch_bounds__(256) void agg_norm(const float* __restrict__ agg,
                                                const float* __restrict__ deg,
                                                _Float16* __restrict__ y, int n) {
  const int i = blockIdx.x * 256 + threadIdx.x;
  if (i >= n) return;
  y[i] = (_Float16)(agg[i] / fmaxf(deg[i >> 8], 1.0f));
}

// ---- Host orchestration --------------------------------------------------
#define CEILDIV(a, b) (((a) + (b) - 1) / (b))

extern "C" void kernel_launch(void* const* d_in, const int* in_sizes, int n_in,
                              void* d_out, int out_size, void* d_ws, size_t ws_size,
                              hipStream_t stream) {
  (void)in_sizes; (void)n_in; (void)out_size; (void)ws_size;
  const int E = 50000, R2 = 460, Rh = 230, T = 4, EG = 200000, L = 150000;

  const float* ent   = (const float*)d_in[0];
  const float* relem = (const float*)d_in[1];
  const float* w1    = (const float*)d_in[2];
  const float* w2    = (const float*)d_in[3];
  const float* biasr = (const float*)d_in[4];
  const float* gam   = (const float*)d_in[5];
  const float* bet   = (const float*)d_in[6];
  const float* linw  = (const float*)d_in[7];
  const float* linb  = (const float*)d_in[8];
  const float* rwx   = (const float*)d_in[9];
  const float* rwh   = (const float*)d_in[10];
  const float* rbx   = (const float*)d_in[11];
  const float* rbh   = (const float*)d_in[12];
  const float* ewx   = (const float*)d_in[13];
  const float* ewh   = (const float*)d_in[14];
  const float* ebx   = (const float*)d_in[15];
  const float* ebh   = (const float*)d_in[16];
  const float* wn    = (const float*)d_in[17];
  const float* wl    = (const float*)d_in[18];
  const int* srcA = (const int*)d_in[19];
  const int* dstA = (const int*)d_in[20];
  const int* etA  = (const int*)d_in[21];
  const int* rteA = (const int*)d_in[22];
  const int* segA = (const int*)d_in[23];

  float* outp = (float*)d_out;
  float* hist = outp;                                   // [T, E, 256]
  float* stat = outp + (size_t)4 * E * 256;             // static_emb [E, 256]
  float* h0f  = outp + (size_t)5 * E * 256;             // h0 [R2, 256]

  // Workspace carve (256B aligned)
  char* cw = (char*)d_ws;
  auto cF = [&](size_t n) { float* p = (float*)cw; cw += ((n * 4 + 255) & ~(size_t)255); return p; };
  auto cH = [&](size_t n) { _Float16* p = (_Float16*)cw; cw += ((n * 2 + 255) & ~(size_t)255); return p; };

  float* muS   = cF(256);
  float* vaS   = cF(256);
  float* sums  = cF((size_t)R2 * 256);
  float* cnt   = cF(R2);
  float* deg   = cF(E);
  float* bigA  = cF((size_t)E * 256);      // agg
  float* bigB  = cF((size_t)E * 256);      // e0pre / curpre / hpre
  float* tmpS  = cF((size_t)R2 * 256);     // r_rel / relpre / h0pre
  float* rel32 = cF((size_t)R2 * 256);     // rel (constant over steps)
  _Float16* halfA   = cH((size_t)E * 256); // xe_f16 / cur_f16
  _Float16* halfB   = cH((size_t)E * 256); // h_f16
  _Float16* halfC   = cH((size_t)E * 256); // aggn_f16
  _Float16* relembH = cH((size_t)Rh * 256);
  _Float16* relcatH = cH((size_t)R2 * 256);
  _Float16* relH    = cH((size_t)R2 * 256);
  _Float16* h0H     = cH((size_t)R2 * 256);
  _Float16* xcatH   = cH((size_t)R2 * 512);
  _Float16* wLin = cH(256 * 256);
  _Float16* w1T  = cH(256 * 256);
  _Float16* w2T  = cH(256 * 256);
  _Float16* wRx  = cH((size_t)768 * 512);
  _Float16* wRh  = cH((size_t)768 * 256);
  _Float16* wEx  = cH((size_t)768 * 256);
  _Float16* wEh  = cH((size_t)768 * 256);
  _Float16* wNT  = cH(256 * 256);
  _Float16* wLT  = cH(256 * 256);

  auto gemmL = [&](const _Float16* A, const _Float16* W, const float* bias,
                   float* C, int M, int N, int K, int accum) {
    const int blocks = CEILDIV(CEILDIV(M, 16), 4);
    gemm_f16<<<blocks, 128, 0, stream>>>(A, W, bias, C, M, N, K, accum);
  };
  auto gruL = [&](const _Float16* X, int Kx, const _Float16* Hf, const float* hprev,
                  const _Float16* Wx, const _Float16* Wh,
                  const float* bx, const float* bh, float* out, int M) {
    const int blocks = CEILDIV(CEILDIV(M, 16), 4);
    gru_fused<<<blocks, 128, 0, stream>>>(X, Kx, Hf, hprev, Wx, Wh, bx, bh, out, M);
  };
  auto zeroL = [&](float* p, int n) {
    zero_f32<<<CEILDIV(n, 256), 256, 0, stream>>>(p, n);
  };

  // ---- Initial: BatchNorm + Linear + normalize; rel embedding pipeline ----
  zeroL(muS, 256);
  zeroL(vaS, 256);
  bn_partial<<<CEILDIV(E, 256), 256, 0, stream>>>(ent, muS, vaS, E);
  bn_finalize<<<1, 256, 0, stream>>>(muS, vaS, 1.0f / (float)E);
  bn_apply<<<CEILDIV(E * 256, 256), 256, 0, stream>>>(ent, muS, vaS, gam, bet, halfA, E * 256);

  conv_f16<<<CEILDIV(256 * 256, 256), 256, 0, stream>>>(linw, wLin, 256 * 256);
  transpose_f16<<<CEILDIV(256 * 256, 256), 256, 0, stream>>>(w1, w1T, 256, 256);
  transpose_f16<<<CEILDIV(256 * 256, 256), 256, 0, stream>>>(w2, w2T, 256, 256);
  conv_f16<<<CEILDIV(768 * 512, 256), 256, 0, stream>>>(rwx, wRx, 768 * 512);
  conv_f16<<<CEILDIV(768 * 256, 256), 256, 0, stream>>>(rwh, wRh, 768 * 256);
  conv_f16<<<CEILDIV(768 * 256, 256), 256, 0, stream>>>(ewx, wEx, 768 * 256);
  conv_f16<<<CEILDIV(768 * 256, 256), 256, 0, stream>>>(ewh, wEh, 768 * 256);
  transpose_f16<<<CEILDIV(256 * 256, 256), 256, 0, stream>>>(wn, wNT, 256, 256);
  transpose_f16<<<CEILDIV(256 * 256, 256), 256, 0, stream>>>(wl, wLT, 256, 256);
  conv_f16<<<CEILDIV(Rh * 256, 256), 256, 0, stream>>>(relem, relembH, Rh * 256);

  // e0 = l2norm(xe @ lin_w.T + lin_b); static = l2norm(e0) (idempotent)
  gemmL(halfA, wLin, linb, bigB, E, 256, 256, 0);
  rownorm<<<E, 256, 0, stream>>>(bigB, stat, halfB, 0);

  // rel = l2norm(concat([rel_emb, rel_emb@w1 + bias_r]) @ w2)
  gemmL(relembH, w1T, biasr, tmpS, Rh, 256, 256, 0);
  concat_rel<<<CEILDIV(R2 * 256, 256), 256, 0, stream>>>(relem, tmpS, relcatH, R2 * 256);
  gemmL(relcatH, w2T, nullptr, tmpS, R2, 256, 256, 0);
  rownorm<<<R2, 256, 0, stream>>>(tmpS, rel32, relH, 0);
  copy_f32<<<CEILDIV(R2 * 256, 256), 256, 0, stream>>>(h0f, rel32, R2 * 256);
  conv_f16<<<CEILDIV(R2 * 256, 256), 256, 0, stream>>>(rel32, h0H, R2 * 256);

  // ---- Recurrent steps ----------------------------------------------------
  for (int t = 0; t < T; ++t) {
    const float* hprev = (t == 0) ? stat : hist + (size_t)(t - 1) * E * 256;
    const int* srcT = srcA + (size_t)t * EG;
    const int* dstT = dstA + (size_t)t * EG;
    const int* etT  = etA  + (size_t)t * EG;
    const int* rteT = rteA + (size_t)t * L;
    const int* segT = segA + (size_t)t * L;

    // segment mean of h[r_to_e] by rel_seg
    zeroL(sums, R2 * 256);
    zeroL(cnt, R2);
    seg_scatter<<<CEILDIV(L * 64, 256), 256, 0, stream>>>(hprev, rteT, segT, sums, L);
    cnt_kernel<<<CEILDIV(L, 256), 256, 0, stream>>>(segT, cnt, L);
    build_xcat<<<CEILDIV(R2 * 512, 256), 256, 0, stream>>>(rel32, sums, cnt, xcatH, R2 * 512);

    // h0 = l2norm(GRU(x_input, h0))
    gruL(xcatH, 512, h0H, h0f, wRx, wRh, rbx, rbh, tmpS, R2);
    rownorm<<<R2, 256, 0, stream>>>(tmpS, h0f, h0H, 0);

    // edge aggregation: agg[d] += h[s] + h0[r], degree norm
    zeroL(bigA, E * 256);
    zeroL(deg, E);
    deg_kernel<<<CEILDIV(EG, 256), 256, 0, stream>>>(dstT, deg, EG);
    edge_scatter<<<CEILDIV(EG * 64, 256), 256, 0, stream>>>(hprev, h0f, srcT, dstT, etT, bigA, EG);
    agg_norm<<<CEILDIV(E * 256, 256), 256, 0, stream>>>(bigA, deg, halfC, E * 256);

    // cur = rrelu(aggn @ w_neighbor + h @ w_loop); l2norm -> f16
    gemmL(halfC, wNT, nullptr, bigB, E, 256, 256, 0);
    gemmL(halfB, wLT, nullptr, bigB, E, 256, 256, 1);
    rownorm<<<E, 256, 0, stream>>>(bigB, nullptr, halfA, 1);

    // h = l2norm(GRU(cur, h)) -> hist[t] (doubles as next-step h)
    gruL(halfA, 256, halfB, hprev, wEx, wEh, ebx, ebh, bigB, E);
    rownorm<<<E, 256, 0, stream>>>(bigB, hist + (size_t)t * E * 256, halfB, 0);
  }
}
